// SelfAttV1_55044300865793
// MI455X (gfx1250) — matrix operand order repaired and checked by reading, not compiled, not run on hardware
//
#include <hip/hip_runtime.h>
#include <hip/hip_bf16.h>

typedef __bf16 bf16;
typedef __attribute__((ext_vector_type(16))) __bf16 v16bf;
typedef __attribute__((ext_vector_type(8)))  __bf16 v8bf;
typedef __attribute__((ext_vector_type(4)))  __bf16 v4bf;
typedef __attribute__((ext_vector_type(8)))  float  v8f;
typedef __attribute__((ext_vector_type(4)))  float  v4f;

#define HIDDEN 1024
#define BM 256
#define BN 128
#define BK 64
#define LDSS 72   // padded LDS row stride in bf16 elems (144B -> conflict-free b128 reads)

union V16U { v16bf v; v8bf h[2]; };

// generic shared pointer -> 32-bit LDS byte offset
__device__ __forceinline__ unsigned lds_off(const void* p) {
    return (unsigned)(unsigned long long)(__attribute__((address_space(3))) const void*)p;
}

// async DMA: LDS[lds+imm] = MEM[g+imm] (16 bytes per lane), tracked by ASYNCcnt
#define ASYNC_B128(ldsoff, gptr, imm)                                          \
    asm volatile("global_load_async_to_lds_b128 %0, %1, off offset:" #imm      \
                 :: "v"(ldsoff), "v"(gptr) : "memory")

__device__ __forceinline__ void wait_async0() {
    asm volatile("s_wait_asynccnt 0x0" ::: "memory");
}

// C[M,N] = (A[M,K] * B[N,K]^T) * scale (+bias). All NT: B rows are output cols.
template<bool OUT_BF16, bool HAS_BIAS>
__global__ __launch_bounds__(256)
void gemm_nt_wmma(const bf16* __restrict__ A, long long sAz, int lda,
                  const bf16* __restrict__ B, long long sBz, int ldb,
                  void* __restrict__ Cv, long long sCz, int ldc,
                  const float* __restrict__ bias, float scale, int K)
{
    __shared__ alignas(16) bf16 Al[2][BM * LDSS];
    __shared__ alignas(16) bf16 Bl[2][BN * LDSS];

    const int tid  = threadIdx.x;
    const int lane = tid & 31;
    const int wave = tid >> 5;
    const int wm   = wave >> 1;          // 0..3 -> 64-row slab
    const int wn   = wave & 1;           // 0..1 -> 64-col slab
    const int tm0  = blockIdx.y * BM;
    const int tn0  = blockIdx.x * BN;

    A += sAz * (long long)blockIdx.z;
    B += sBz * (long long)blockIdx.z;

    const v8f zero = {0.f,0.f,0.f,0.f,0.f,0.f,0.f,0.f};
    v8f acc[4][4];
    #pragma unroll
    for (int m = 0; m < 4; m++)
        #pragma unroll
        for (int n = 0; n < 4; n++)
            acc[m][n] = zero;

    // staging mapping: A: one full 64-half row per thread; B: half row per thread
    const int arow = tid;                 // 0..255
    const int brow = tid >> 1;            // 0..127
    const int bseg = (tid & 1) * 32;      // 0 / 32 halves

    const unsigned la0 = lds_off(&Al[0][arow * LDSS]);
    const unsigned lb0 = lds_off(&Bl[0][brow * LDSS + bseg]);
    const unsigned lbufsz_a = (unsigned)(BM * LDSS * sizeof(bf16));
    const unsigned lbufsz_b = (unsigned)(BN * LDSS * sizeof(bf16));
    const bf16* agbase = A + (long long)(tm0 + arow) * lda;
    const bf16* bgbase = B + (long long)(tn0 + brow) * ldb + bseg;

    const int fr = lane & 15;
    const int hi = lane >> 4;

    // ---- issue async stage of tile kt into buffer `buf` ----
    auto stage = [&](int buf, int kt) {
        const unsigned la = la0 + buf * lbufsz_a;
        const bf16* ag = agbase + kt;
        ASYNC_B128(la, ag, 0);   ASYNC_B128(la, ag, 16);
        ASYNC_B128(la, ag, 32);  ASYNC_B128(la, ag, 48);
        ASYNC_B128(la, ag, 64);  ASYNC_B128(la, ag, 80);
        ASYNC_B128(la, ag, 96);  ASYNC_B128(la, ag, 112);
        const unsigned lb = lb0 + buf * lbufsz_b;
        const bf16* bg = bgbase + kt;
        ASYNC_B128(lb, bg, 0);   ASYNC_B128(lb, bg, 16);
        ASYNC_B128(lb, bg, 32);  ASYNC_B128(lb, bg, 48);
    };

    stage(0, 0);
    wait_async0();
    __syncthreads();

    int buf = 0;
    for (int kt = 0; kt < K; kt += BK) {
        if (kt + BK < K) stage(buf ^ 1, kt + BK);   // overlap with compute below

        #pragma unroll
        for (int ks = 0; ks < BK; ks += 32) {
            V16U af[4], bfg[4];
            #pragma unroll
            for (int m = 0; m < 4; m++) {
                const int r  = wm * 64 + m * 16 + fr;
                const int kb = ks + hi * 8;               // bf16 A 16x32 layout
                af[m].h[0] = *(const v8bf*)&Al[buf][r * LDSS + kb];
                af[m].h[1] = *(const v8bf*)&Al[buf][r * LDSS + kb + 16];
            }
            #pragma unroll
            for (int n = 0; n < 4; n++) {
                const int c  = wn * 64 + n * 16 + fr;
                const int kb = ks + hi * 8;               // bf16 B 32x16 layout
                bfg[n].h[0] = *(const v8bf*)&Bl[buf][c * LDSS + kb];
                bfg[n].h[1] = *(const v8bf*)&Bl[buf][c * LDSS + kb + 16];
            }
            #pragma unroll
            for (int n = 0; n < 4; n++)
                #pragma unroll
                for (int m = 0; m < 4; m++)
                    acc[m][n] = __builtin_amdgcn_wmma_f32_16x16x32_bf16(
                        false, af[m].v, false, bfg[n].v,
                        (short)0, acc[m][n], false, false);
        }

        wait_async0();     // staged tile for next iter has landed in LDS
        __syncthreads();   // all waves done reading `buf`, writes visible
        buf ^= 1;
    }

    // ---- epilogue: scale (+bias), store f32 or bf16 ----
    const long long cz = sCz * (long long)blockIdx.z;
    #pragma unroll
    for (int n = 0; n < 4; n++) {
        const int gc = tn0 + wn * 64 + n * 16 + fr;
        const float bv = HAS_BIAS ? bias[gc] : 0.f;
        #pragma unroll
        for (int m = 0; m < 4; m++) {
            const int gr0 = tm0 + wm * 64 + m * 16 + hi * 8;
            #pragma unroll
            for (int e = 0; e < 8; e++) {
                const float val = acc[m][n][e] * scale + bv;
                const long long off = cz + (long long)(gr0 + e) * ldc + gc;
                if (OUT_BF16) ((bf16*)Cv)[off] = (bf16)val;
                else          ((float*)Cv)[off] = val;
            }
        }
    }
}

// row softmax: 2048 f32 in -> 2048 bf16 out, one block per row
__global__ __launch_bounds__(256)
void softmax_row(const float* __restrict__ S, bf16* __restrict__ P)
{
    const int N = 2048;
    const long long row = blockIdx.x;
    const float* s = S + row * N;
    bf16* p = P + row * N;
    const int t = threadIdx.x, lane = t & 31, wv = t >> 5;
    __shared__ float red[8];

    float v[8];
    float mx = -3.402823e38f;
    #pragma unroll
    for (int i = 0; i < 8; i++) { v[i] = s[t + (i << 8)]; mx = fmaxf(mx, v[i]); }
    #pragma unroll
    for (int o = 16; o > 0; o >>= 1) mx = fmaxf(mx, __shfl_xor(mx, o, 32));
    if (lane == 0) red[wv] = mx;
    __syncthreads();
    mx = red[0];
    #pragma unroll
    for (int w = 1; w < 8; w++) mx = fmaxf(mx, red[w]);

    float sum = 0.f;
    #pragma unroll
    for (int i = 0; i < 8; i++) { v[i] = __expf(v[i] - mx); sum += v[i]; }
    #pragma unroll
    for (int o = 16; o > 0; o >>= 1) sum += __shfl_xor(sum, o, 32);
    __syncthreads();
    if (lane == 0) red[wv] = sum;
    __syncthreads();
    sum = 0.f;
    #pragma unroll
    for (int w = 0; w < 8; w++) sum += red[w];
    const float inv = __builtin_amdgcn_rcpf(sum);
    #pragma unroll
    for (int i = 0; i < 8; i++) p[t + (i << 8)] = (bf16)(v[i] * inv);
}

__global__ __launch_bounds__(256)
void cast_f32_bf16(const float* __restrict__ in, bf16* __restrict__ out)
{
    const long long i = ((long long)blockIdx.x * 256 + threadIdx.x) * 4;
    v4f f = *(const v4f*)(in + i);
    v4bf o;
    #pragma unroll
    for (int j = 0; j < 4; j++) o[j] = (bf16)f[j];
    *(v4bf*)(out + i) = o;
}

// Wt[n][k] = W_sel[k][n%1024] (bf16), n in [0,3072): Wq|Wk|Wv transposed
__global__ __launch_bounds__(256)
void pack_wt(const float* __restrict__ Wq, const float* __restrict__ Wk,
             const float* __restrict__ Wv, bf16* __restrict__ Wt)
{
    __shared__ float tile[32][33];
    const int tx = threadIdx.x & 31;
    const int ty = threadIdx.x >> 5;      // 0..7
    const int nb = blockIdx.x * 32;
    const int kb = blockIdx.y * 32;
    const int sel = nb >> 10;
    const float* W = sel == 0 ? Wq : (sel == 1 ? Wk : Wv);
    const int nl = (nb & 1023) + tx;
    #pragma unroll
    for (int j = 0; j < 4; j++) {
        const int k = kb + ty + j * 8;
        tile[ty + j * 8][tx] = W[k * HIDDEN + nl];
    }
    __syncthreads();
    #pragma unroll
    for (int j = 0; j < 4; j++) {
        const int n = nb + ty + j * 8;
        const int k = kb + tx;
        Wt[(long long)n * HIDDEN + k] = (bf16)tile[tx][ty + j * 8];
    }
}

// Vt[b][d][s] = V[b][s][d]; V lives in packed qkv at column offset 2048
__global__ __launch_bounds__(256)
void transpose_v(const bf16* __restrict__ qkv, bf16* __restrict__ Vt)
{
    __shared__ bf16 tile[32][33];
    const int tx = threadIdx.x & 31;
    const int ty = threadIdx.x >> 5;
    const int d0 = blockIdx.x * 32;       // 0..1023
    const int s0 = blockIdx.y * 32;       // 0..2047
    const long long b = blockIdx.z;
    const bf16* V = qkv + b * 2048LL * 3072 + 2048;
    bf16* vt = Vt + b * 1024LL * 2048;
    #pragma unroll
    for (int j = 0; j < 4; j++)
        tile[ty + j * 8][tx] = V[(long long)(s0 + ty + j * 8) * 3072 + d0 + tx];
    __syncthreads();
    #pragma unroll
    for (int j = 0; j < 4; j++)
        vt[(long long)(d0 + ty + j * 8) * 2048 + s0 + tx] = tile[tx][ty + j * 8];
}

__global__ __launch_bounds__(256)
void pack_bias(const float* __restrict__ bq, const float* __restrict__ bk,
               const float* __restrict__ bv, float* __restrict__ out)
{
    const int i = blockIdx.x * 256 + threadIdx.x;
    if (i < 1024)      out[i] = bq[i];
    else if (i < 2048) out[i] = bk[i - 1024];
    else               out[i] = bv[i - 2048];
}

extern "C" void kernel_launch(void* const* d_in, const int* in_sizes, int n_in,
                              void* d_out, int out_size, void* d_ws, size_t ws_size,
                              hipStream_t stream)
{
    (void)in_sizes; (void)n_in; (void)out_size; (void)ws_size;
    const float* x  = (const float*)d_in[0];
    const float* Wq = (const float*)d_in[1];
    const float* bq = (const float*)d_in[2];
    const float* Wk = (const float*)d_in[3];
    const float* bk = (const float*)d_in[4];
    const float* Wv = (const float*)d_in[5];
    const float* bv = (const float*)d_in[6];
    float* out = (float*)d_out;

    const long long M = 4LL * 2048;   // 8192 rows
    char* w = (char*)d_ws;
    bf16*  xb   = (bf16*)w;  w += M * HIDDEN * sizeof(bf16);             // 16 MB
    bf16*  wt   = (bf16*)w;  w += 3LL * HIDDEN * HIDDEN * sizeof(bf16);  // 6 MB
    float* bias = (float*)w; w += 3LL * HIDDEN * sizeof(float);          // 12 KB
    bf16*  qkv  = (bf16*)w;  w += M * 3 * HIDDEN * sizeof(bf16);         // 48 MB
    float* sc   = (float*)w; w += 4LL * 2048 * 2048 * sizeof(float);     // 64 MB
    bf16*  p    = (bf16*)w;  w += 4LL * 2048 * 2048 * sizeof(bf16);      // 32 MB
    bf16*  vt   = (bf16*)w;                                              // 16 MB

    cast_f32_bf16<<<8192, 256, 0, stream>>>(x, xb);
    pack_wt<<<dim3(96, 32), 256, 0, stream>>>(Wq, Wk, Wv, wt);
    pack_bias<<<12, 256, 0, stream>>>(bq, bk, bv, bias);

    // QKV = xb @ Wt^T + bias  -> bf16 [8192][3072]
    gemm_nt_wmma<true, true><<<dim3(3072 / BN, 8192 / BM, 1), 256, 0, stream>>>(
        xb, 0, HIDDEN, wt, 0, HIDDEN, qkv, 0, 3 * HIDDEN, bias, 1.0f, HIDDEN);

    // scores = (Q @ K^T) / 32  -> f32 [4][2048][2048]
    gemm_nt_wmma<false, false><<<dim3(2048 / BN, 2048 / BM, 4), 256, 0, stream>>>(
        qkv,          2048LL * 3 * HIDDEN, 3 * HIDDEN,
        qkv + HIDDEN, 2048LL * 3 * HIDDEN, 3 * HIDDEN,
        sc, 2048LL * 2048, 2048, nullptr, 0.03125f, HIDDEN);

    softmax_row<<<4 * 2048, 256, 0, stream>>>(sc, p);

    // Vt[b][d][s]
    transpose_v<<<dim3(32, 64, 4), 256, 0, stream>>>(qkv, vt);

    // out = P @ Vt^T -> f32 [4][2048][1024]
    gemm_nt_wmma<false, false><<<dim3(HIDDEN / BN, 2048 / BM, 4), 256, 0, stream>>>(
        p, 2048LL * 2048, 2048,
        vt, 1024LL * 2048, 2048,
        out, 2048LL * HIDDEN, HIDDEN, nullptr, 1.0f, 2048);
}
